// LEMODE2_73254962200837
// MI455X (gfx1250) — matrix operand order repaired and checked by reading, not compiled, not run on hardware
//
#include <hip/hip_runtime.h>
#include <hip/hip_bf16.h>

#define NINP 256
#define NHID 256
#define NOUT 256
#define T_STEPS 128
#define B_ROWS 256
#define DIV_STEPS 10

typedef __bf16 bf16;
typedef __attribute__((ext_vector_type(16))) __bf16 v16bf;
typedef __attribute__((ext_vector_type(8)))  __bf16 v8bf;
typedef __attribute__((ext_vector_type(8)))  float  v8f;

struct Params {
    // inputs (fp32, from d_in)
    const float *input, *dt;
    const float *Wi_f, *bi, *Wh_f, *bh, *Wz_f, *bz, *Wdt, *bdt, *Wc_f, *bc, *Ws_f, *bs;
    // bf16 weight copies + activations in workspace
    bf16 *Wi, *Wh, *Wz, *Wc, *Ws;
    bf16 *x_bf, *noise_bf, *yh_bf;
    bf16 *y_bf, *z_bf;          // bf16 shadows of the f32 recurrent state
    float *ti, *th, *sb, *y, *z;
    float *out;
    unsigned *bar;
};

__device__ __forceinline__ float sigmoidf_(float x) { return 1.0f / (1.0f + __expf(-x)); }

// Threefry-2x32, 20 rounds (counter-based RNG in the spirit of JAX's).
__device__ __forceinline__ void threefry2x32(unsigned k0, unsigned k1,
                                             unsigned c0, unsigned c1,
                                             unsigned &o0, unsigned &o1) {
    const unsigned R0[4] = {13, 15, 26, 6};
    const unsigned R1[4] = {17, 29, 16, 24};
    unsigned ks[3] = {k0, k1, 0x1BD11BDAu ^ k0 ^ k1};
    unsigned x0 = c0 + k0, x1 = c1 + k1;
#pragma unroll
    for (int g = 0; g < 5; ++g) {
        const unsigned *r = (g & 1) ? R1 : R0;
#pragma unroll
        for (int i = 0; i < 4; ++i) {
            x0 += x1;
            x1 = (x1 << r[i]) | (x1 >> (32u - r[i]));
            x1 ^= x0;
        }
        x0 += ks[(g + 1) % 3];
        x1 += ks[(g + 2) % 3] + (unsigned)(g + 1);
    }
    o0 = x0; o1 = x1;
}

// Fill noise_bf[B_ROWS x 2*NHID] with N(0,1) samples for scan step `sidx`
// (folded key, per-pair counter, Box-Muller).
__device__ __forceinline__ void noise_fill(const Params &p, unsigned sidx, int g, int tot) {
    unsigned fk0, fk1;
    threefry2x32(0u, 1234u, 0u, sidx, fk0, fk1);           // fold_in(key, step)
    for (int pi = g; pi < B_ROWS * NHID; pi += tot) {      // one pair per item
        unsigned r0, r1;
        threefry2x32(fk0, fk1, (unsigned)pi, 0x9E3779B9u, r0, r1);
        float u1 = ((r0 >> 8) + 1u) * (1.0f / 16777217.0f);
        float u2 = (r1 >> 8) * (1.0f / 16777216.0f);
        float m = __fsqrt_rn(-2.0f * __logf(u1));
        float sn, cs;
        __sincosf(6.28318530718f * u2, &sn, &cs);
        p.noise_bf[2 * pi]     = (bf16)(m * cs);
        p.noise_bf[2 * pi + 1] = (bf16)(m * sn);
    }
}

// Grid-wide barrier (monotonic counter, reset by the init kernel each launch).
__device__ __forceinline__ void gbar(unsigned *cnt, unsigned &gen, unsigned nblk) {
    __syncthreads();
    __threadfence();
    ++gen;
    if (threadIdx.x == 0) {
        __hip_atomic_fetch_add(cnt, 1u, __ATOMIC_RELEASE, __HIP_MEMORY_SCOPE_AGENT);
        const unsigned target = gen * nblk;
        while (__hip_atomic_load(cnt, __ATOMIC_ACQUIRE, __HIP_MEMORY_SCOPE_AGENT) < target)
            __builtin_amdgcn_s_sleep(8);
    }
    __syncthreads();
    __threadfence();
}

// One 16x16 output tile of  Out = A(MxK) * W(NxK)^T, bf16 WMMA, f32 accumulate.
// A layout: lane holds row=lane&15, 16 contiguous K values starting at
// k + (lane>=16 ? 16 : 0).  B: lane holds column n=lane&15, 16
// contiguous K values starting at k + (lane>=16 ? 16 : 0).
template <bool A_IS_F32>
__device__ __forceinline__ v8f gemm_tile(const void *Abase, int lda,
                                         const bf16 *W, int ldb, int K,
                                         int m0, int n0, int lane) {
    const int half = lane >> 4;
    const int r = lane & 15;
    const bf16 *brow = W + (size_t)(n0 + r) * ldb + half * 16;
    v8f acc = {};
    if (A_IS_F32) {
        const float *arow = (const float *)Abase + (size_t)(m0 + r) * lda + half * 16;
        for (int k = 0; k < K; k += 32) {
            v16bf a;
#pragma unroll
            for (int i = 0; i < 8; ++i) {
                a[i]     = (bf16)arow[k + i];
                a[i + 8] = (bf16)arow[k + 8 + i];
            }
            v16bf b = *(const v16bf *)(brow + k);
            if (k + 32 < K) __builtin_prefetch((const void *)(brow + k + 32), 0, 0);
            acc = __builtin_amdgcn_wmma_f32_16x16x32_bf16(false, a, false, b,
                                                          (short)0, acc, false, false);
        }
    } else {
        const bf16 *arow = (const bf16 *)Abase + (size_t)(m0 + r) * lda + half * 16;
        for (int k = 0; k < K; k += 32) {
            v8bf a0 = *(const v8bf *)(arow + k);
            v8bf a1 = *(const v8bf *)(arow + k + 8);
            v16bf a;
#pragma unroll
            for (int i = 0; i < 8; ++i) { a[i] = a0[i]; a[i + 8] = a1[i]; }
            v16bf b = *(const v16bf *)(brow + k);
            if (k + 32 < K) __builtin_prefetch((const void *)(brow + k + 32), 0, 0);
            acc = __builtin_amdgcn_wmma_f32_16x16x32_bf16(false, a, false, b,
                                                          (short)0, acc, false, false);
        }
    }
    return acc;
}

// D layout (32-bit C/D 16x16): VGPR i holds row m0+i (lanes 0-15) / m0+i+8
// (lanes 16-31), column n0+(lane&15).
__device__ __forceinline__ void store_tile(float *O, int ldo, int m0, int n0,
                                           int lane, v8f acc, const float *bias,
                                           bool do_tanh) {
    const int half = lane >> 4, r = lane & 15, n = n0 + r;
    const float bv = bias[n];
#pragma unroll
    for (int i = 0; i < 8; ++i) {
        float v = acc[i] + bv;
        if (do_tanh) v = tanhf(v);
        O[(size_t)(m0 + i + half * 8) * ldo + n] = v;
    }
}

// ---------------- Kernel 1: init (reset barrier, cast weights, zero state, noise0)
__global__ void lem_init(Params p) {
    const int g = blockIdx.x * blockDim.x + threadIdx.x;
    const int tot = gridDim.x * blockDim.x;
    if (g == 0) *p.bar = 0u;
    for (int i = g; i < 4 * NHID * NINP; i += tot) p.Wi[i] = (bf16)p.Wi_f[i];
    for (int i = g; i < 3 * NHID * NHID; i += tot) p.Wh[i] = (bf16)p.Wh_f[i];
    for (int i = g; i < NHID * NHID; i += tot)     p.Wz[i] = (bf16)p.Wz_f[i];
    for (int i = g; i < NOUT * NHID; i += tot)     p.Wc[i] = (bf16)p.Wc_f[i];
    for (int i = g; i < 4 * NHID * NHID; i += tot) p.Ws[i] = (bf16)p.Ws_f[i];
    for (int i = g; i < B_ROWS * NHID; i += tot) {
        p.y[i] = 0.0f; p.z[i] = 0.0f;
        p.y_bf[i] = (bf16)0.0f; p.z_bf[i] = (bf16)0.0f;
    }
    noise_fill(p, 0u, g, tot);
}

// ---------------- Kernel 2: persistent sequential scan
__global__ void __launch_bounds__(256) lem_persist(Params p) {
    const int tid = threadIdx.x;
    const int lane = tid & 31;
    const int gthreads = gridDim.x * blockDim.x;
    const int gtid = blockIdx.x * blockDim.x + tid;
    const int wv = gtid >> 5;
    const int nw = gthreads >> 5;
    unsigned gen = 0;

    const float w00 = p.Wdt[0], bd0 = p.bdt[0];
    const float w10 = p.Wdt[1], bd1 = p.bdt[1];

    for (int t = 0; t < T_STEPS; ++t) {
        for (int d = 0; d < DIV_STEPS; ++d) {
            // ---- P1: ti = x*Wi^T+bi ; th = y*Wh^T+bh ; sb = tanh(noise*Ws^T+bs)
            const int TI_T = (B_ROWS / 16) * ((4 * NHID) / 16);  // 1024 tiles
            const int TH_T = (B_ROWS / 16) * ((3 * NHID) / 16);  // 768
            const int S_T  = (B_ROWS / 16) * ((2 * NHID) / 16);  // 512
            for (int tile = wv; tile < TI_T + TH_T + S_T; tile += nw) {
                if (tile < TI_T) {
                    int bm = tile >> 6, nt = tile & 63;
                    v8f acc;
                    if (d == 0)  // inner-scan carry x starts as input[t] (f32)
                        acc = gemm_tile<true>(p.input + (size_t)t * B_ROWS * NINP, NINP,
                                              p.Wi, NINP, NINP, bm * 16, nt * 16, lane);
                    else
                        acc = gemm_tile<false>(p.x_bf, NHID,
                                               p.Wi, NHID, NHID, bm * 16, nt * 16, lane);
                    store_tile(p.ti, 4 * NHID, bm * 16, nt * 16, lane, acc, p.bi, false);
                } else if (tile < TI_T + TH_T) {
                    int t2 = tile - TI_T;
                    int bm = t2 / 48, nt = t2 % 48;
                    v8f acc = gemm_tile<false>(p.y_bf, NHID, p.Wh, NHID, NHID,
                                               bm * 16, nt * 16, lane);
                    store_tile(p.th, 3 * NHID, bm * 16, nt * 16, lane, acc, p.bh, false);
                } else {
                    int t3 = tile - TI_T - TH_T;
                    int bm = t3 >> 5, nt = t3 & 31;
                    v8f acc = gemm_tile<false>(p.noise_bf, 2 * NHID, p.Ws, 2 * NHID,
                                               2 * NHID, bm * 16, nt * 16, lane);
                    store_tile(p.sb, 2 * NHID, bm * 16, nt * 16, lane, acc, p.bs, true);
                }
            }
            gbar(p.bar, gen, gridDim.x);

            // ---- P2: z = (1-ms_dt)*z + ms_dt*tanh(i_y + h_y)   (+ bf16 shadow)
            for (int idx = gtid; idx < B_ROWS * NHID; idx += gthreads) {
                int b = idx >> 8, h = idx & 255;
                float delta = p.dt[t * B_ROWS + b] * 0.1f;
                float td2 = sigmoidf_(delta * w10 + bd1);
                float ms  = td2 * sigmoidf_(p.ti[b * 1024 + 256 + h] +
                                            p.th[b * 768 + 256 + h]);
                float cnd = tanhf(p.ti[b * 1024 + 768 + h] + p.th[b * 768 + 512 + h]);
                float zv = (1.0f - ms) * p.z[idx] + ms * cnd;
                p.z[idx] = zv;
                p.z_bf[idx] = (bf16)zv;
            }
            gbar(p.bar, gen, gridDim.x);

            // ---- P3: y = (1-ms_dt_bar)*y + ms_dt_bar*tanh(z*Wz^T + bz + i_z)
            for (int tile = wv; tile < 256; tile += nw) {
                int m0 = (tile >> 4) * 16, n0 = (tile & 15) * 16;
                v8f acc = gemm_tile<false>(p.z_bf, NHID, p.Wz, NHID, NHID, m0, n0, lane);
                int half = lane >> 4, r = lane & 15, n = n0 + r;
                float bz = p.bz[n];
#pragma unroll
                for (int i = 0; i < 8; ++i) {
                    int row = m0 + i + half * 8;
                    float delta = p.dt[t * B_ROWS + row] * 0.1f;
                    float td1 = sigmoidf_(delta * w00 + bd0);
                    float msb = td1 * sigmoidf_(p.ti[row * 1024 + n] + p.th[row * 768 + n]);
                    float u = acc[i] + bz + p.ti[row * 1024 + 512 + n];
                    float yo = p.y[row * NHID + n];
                    float yv = (1.0f - msb) * yo + msb * tanhf(u);
                    p.y[row * NHID + n] = yv;
                    p.y_bf[row * NHID + n] = (bf16)yv;
                }
            }
            gbar(p.bar, gen, gridDim.x);

            // ---- P4: x = y*Wc^T + bc  (pre-noise y; skip on last inner step)
            if (d != DIV_STEPS - 1) {
                for (int tile = wv; tile < 256; tile += nw) {
                    int m0 = (tile >> 4) * 16, n0 = (tile & 15) * 16;
                    v8f acc = gemm_tile<false>(p.y_bf, NHID, p.Wc, NHID, NHID,
                                               m0, n0, lane);
                    int half = lane >> 4, r = lane & 15, n = n0 + r;
                    float bc = p.bc[n];
#pragma unroll
                    for (int i = 0; i < 8; ++i) {
                        int row = m0 + i + half * 8;
                        p.x_bf[row * NHID + n] = (bf16)(acc[i] + bc);
                    }
                }
                gbar(p.bar, gen, gridDim.x);
            }

            // ---- P5: y += ny*delta ; z += nz*delta ; shadows ; y_hid ; next noise
            for (int idx = gtid; idx < B_ROWS * NHID; idx += gthreads) {
                int b = idx >> 8, h = idx & 255;
                float delta = p.dt[t * B_ROWS + b] * 0.1f;
                float yv = p.y[idx] + p.sb[b * 512 + h] * delta;        // YCOEFF=1
                float zv = p.z[idx] + p.sb[b * 512 + 256 + h] * delta;  // ZCOEFF=1
                p.y[idx] = yv;  p.y_bf[idx] = (bf16)yv;
                p.z[idx] = zv;  p.z_bf[idx] = (bf16)zv;
                if (d == DIV_STEPS - 1)
                    p.yh_bf[(size_t)t * B_ROWS * NHID + idx] = (bf16)yv;
            }
            if (!(t == T_STEPS - 1 && d == DIV_STEPS - 1))
                noise_fill(p, (unsigned)(t * DIV_STEPS + d + 1), gtid, gthreads);
            gbar(p.bar, gen, gridDim.x);
        }
    }
}

// ---------------- Kernel 3: out = y_hidds * Wc^T + bc  (fully parallel WMMA GEMM)
__global__ void __launch_bounds__(256) lem_out(Params p) {
    const int lane = threadIdx.x & 31;
    const int wv = (blockIdx.x * blockDim.x + threadIdx.x) >> 5;
    const int nw = (gridDim.x * blockDim.x) >> 5;
    const int NT = (T_STEPS * B_ROWS / 16) * (NOUT / 16);
    for (int tile = wv; tile < NT; tile += nw) {
        int bm = tile >> 4, nt = tile & 15;
        v8f acc = gemm_tile<false>(p.yh_bf, NHID, p.Wc, NHID, NHID,
                                   bm * 16, nt * 16, lane);
        store_tile(p.out, NOUT, bm * 16, nt * 16, lane, acc, p.bc, false);
    }
}

extern "C" void kernel_launch(void* const* d_in, const int* in_sizes, int n_in,
                              void* d_out, int out_size, void* d_ws, size_t ws_size,
                              hipStream_t stream) {
    (void)in_sizes; (void)n_in; (void)out_size; (void)ws_size;
    Params p{};
    p.input = (const float*)d_in[0];
    p.dt    = (const float*)d_in[1];
    p.Wi_f  = (const float*)d_in[2];  p.bi  = (const float*)d_in[3];
    p.Wh_f  = (const float*)d_in[4];  p.bh  = (const float*)d_in[5];
    p.Wz_f  = (const float*)d_in[6];  p.bz  = (const float*)d_in[7];
    p.Wdt   = (const float*)d_in[8];  p.bdt = (const float*)d_in[9];
    p.Wc_f  = (const float*)d_in[10]; p.bc  = (const float*)d_in[11];
    p.Ws_f  = (const float*)d_in[12]; p.bs  = (const float*)d_in[13];
    p.out   = (float*)d_out;

    char* w = (char*)d_ws;
    size_t off = 0;
    auto take = [&](size_t bytes) -> char* {
        char* r = w + off;
        off = (off + bytes + 63) & ~(size_t)63;
        return r;
    };
    p.bar      = (unsigned*)take(64);
    p.Wi       = (bf16*)take((size_t)4 * NHID * NINP * 2);
    p.Wh       = (bf16*)take((size_t)3 * NHID * NHID * 2);
    p.Wz       = (bf16*)take((size_t)NHID * NHID * 2);
    p.Wc       = (bf16*)take((size_t)NOUT * NHID * 2);
    p.Ws       = (bf16*)take((size_t)4 * NHID * NHID * 2);
    p.x_bf     = (bf16*)take((size_t)B_ROWS * NHID * 2);
    p.noise_bf = (bf16*)take((size_t)B_ROWS * 2 * NHID * 2);
    p.yh_bf    = (bf16*)take((size_t)T_STEPS * B_ROWS * NHID * 2);
    p.y_bf     = (bf16*)take((size_t)B_ROWS * NHID * 2);
    p.z_bf     = (bf16*)take((size_t)B_ROWS * NHID * 2);
    p.ti       = (float*)take((size_t)B_ROWS * 4 * NHID * 4);
    p.th       = (float*)take((size_t)B_ROWS * 3 * NHID * 4);
    p.sb       = (float*)take((size_t)B_ROWS * 2 * NHID * 4);
    p.y        = (float*)take((size_t)B_ROWS * NHID * 4);
    p.z        = (float*)take((size_t)B_ROWS * NHID * 4);

    hipLaunchKernelGGL(lem_init,    dim3(256),  dim3(256), 0, stream, p);
    hipLaunchKernelGGL(lem_persist, dim3(64),   dim3(256), 0, stream, p);
    hipLaunchKernelGGL(lem_out,     dim3(2048), dim3(256), 0, stream, p);
}
